// E2E_29051158790191
// MI455X (gfx1250) — matrix operand order repaired and checked by reading, not compiled, not run on hardware
//
#include <hip/hip_runtime.h>
#include <hip/hip_bf16.h>
#include <math.h>

// ---------------- types for CDNA5 WMMA ----------------
typedef __bf16 bf16_t;
typedef __attribute__((ext_vector_type(16))) __bf16 v16bf;
typedef __attribute__((ext_vector_type(8)))  __bf16 v8bf;
typedef __attribute__((ext_vector_type(8)))  float  v8f;

#define NB 32      // batch
#define NT 512     // encoder time
#define NENC 512
#define NL 100
#define NV 1000
#define NE 512
#define NH 1024
#define NATT 512
#define NC 10
#define NK 201
#define NOD 512
#define NVPAD 1008 // 63 tiles of 16

__device__ __forceinline__ v8f wmma_bf16(v16bf a, v16bf b, v8f c) {
  // D = A(16x32 bf16) x B(32x16 bf16) + C(16x16 f32)
  return __builtin_amdgcn_wmma_f32_16x16x32_bf16(false, a, false, b, (short)0, c,
                                                 false, false);
}

union ABfrag { v16bf v; v8bf h[2]; };

// ---------------------------------------------------------------------------
// Generic GEMM:  C[m,n] = sum_k A[m,k]*B[n,k] (+ sum_k2 A2[m,k]*B2[n,k]) + bias[n]
// A,A2: bf16 row-major [M x lda]; B,B2: bf16 row-major [N x ldb] (i.e. B^T GEMM)
// Block = 256 threads = 8 waves; wave -> one 16-wide N tile, 32 rows of M.
// gridDim.y tiles M in steps of 32. K, K2 multiples of 32.
// ---------------------------------------------------------------------------
__global__ void gemm32_bf16(const bf16_t* __restrict__ A, const bf16_t* __restrict__ B,
                            int lda, int ldb, int K,
                            const bf16_t* __restrict__ A2, const bf16_t* __restrict__ B2,
                            int lda2, int ldb2, int K2,
                            const float* __restrict__ bias,
                            float* __restrict__ C, long ldc, int Nreal, int Ntiles)
{
  const int wv    = threadIdx.x >> 5;
  const int lane  = threadIdx.x & 31;
  const int ntile = blockIdx.x * 8 + wv;
  if (ntile >= Ntiles) return;
  const long mbase = (long)blockIdx.y * 32;
  const int  half  = lane >> 4;   // which K-half this lane holds
  const int  l16   = lane & 15;   // M row (A) / N col (B) within tile

  v8f acc0 = {0.f,0.f,0.f,0.f,0.f,0.f,0.f,0.f};
  v8f acc1 = {0.f,0.f,0.f,0.f,0.f,0.f,0.f,0.f};

  {
    const bf16_t* a0 = A + (mbase + l16)      * (long)lda + half * 16;
    const bf16_t* a1 = A + (mbase + 16 + l16) * (long)lda + half * 16;
    const bf16_t* bp = B + ((long)ntile * 16 + l16) * (long)ldb + half * 16;
    for (int kt = 0; kt < K; kt += 32) {
      ABfrag fa0, fa1;
      fa0.h[0] = *(const v8bf*)(a0 + kt);
      fa0.h[1] = *(const v8bf*)(a0 + kt + 8);
      fa1.h[0] = *(const v8bf*)(a1 + kt);
      fa1.h[1] = *(const v8bf*)(a1 + kt + 8);
      v16bf bv = *(const v16bf*)(bp + kt);
      acc0 = wmma_bf16(fa0.v, bv, acc0);
      acc1 = wmma_bf16(fa1.v, bv, acc1);
    }
  }
  if (K2 > 0) {
    const bf16_t* a0 = A2 + (mbase + l16)      * (long)lda2 + half * 16;
    const bf16_t* a1 = A2 + (mbase + 16 + l16) * (long)lda2 + half * 16;
    const bf16_t* bp = B2 + ((long)ntile * 16 + l16) * (long)ldb2 + half * 16;
    for (int kt = 0; kt < K2; kt += 32) {
      ABfrag fa0, fa1;
      fa0.h[0] = *(const v8bf*)(a0 + kt);
      fa0.h[1] = *(const v8bf*)(a0 + kt + 8);
      fa1.h[0] = *(const v8bf*)(a1 + kt);
      fa1.h[1] = *(const v8bf*)(a1 + kt + 8);
      v16bf bv = *(const v16bf*)(bp + kt);
      acc0 = wmma_bf16(fa0.v, bv, acc0);
      acc1 = wmma_bf16(fa1.v, bv, acc1);
    }
  }

  const int n = ntile * 16 + l16;
  if (n < Nreal) {
    const float bs = bias ? bias[n] : 0.0f;
    const long r0 = mbase + half * 8;
#pragma unroll
    for (int r = 0; r < 8; ++r) {
      C[(r0 + r)      * ldc + n] = acc0[r] + bs;
      C[(r0 + 16 + r) * ldc + n] = acc1[r] + bs;
    }
  }
}

// ---------------- helpers ----------------
__global__ void cvt_f32_bf16(const float* __restrict__ src, bf16_t* __restrict__ dst, long n) {
  long i = (long)blockIdx.x * blockDim.x + threadIdx.x;
  long stride = (long)gridDim.x * blockDim.x;
  for (; i < n; i += stride) dst[i] = (bf16_t)src[i];
}

__global__ void cvt_wout_pad(const float* __restrict__ src, bf16_t* __restrict__ dst) {
  long i = (long)blockIdx.x * blockDim.x + threadIdx.x;
  long stride = (long)gridDim.x * blockDim.x;
  const long n = (long)NVPAD * 1536;
  for (; i < n; i += stride) {
    int r = (int)(i / 1536), c = (int)(i % 1536);
    dst[i] = (r < NV) ? (bf16_t)src[(long)r * 1536 + c] : (bf16_t)0.0f;
  }
}

__global__ void bias_comb(const float* __restrict__ a, const float* __restrict__ b,
                          float* __restrict__ o) {
  int i = blockIdx.x * 256 + threadIdx.x;
  if (i < 4 * NH) o[i] = a[i] + b[i];
}

__global__ void zero_f32(float* p, long n) {
  long i = (long)blockIdx.x * blockDim.x + threadIdx.x;
  long stride = (long)gridDim.x * blockDim.x;
  for (; i < n; i += stride) p[i] = 0.f;
}

__global__ void init_w(const int* __restrict__ enc_len, float* __restrict__ w) {
  int i = blockIdx.x * 256 + threadIdx.x;        // B*T
  if (i >= NB * NT) return;
  int b = i >> 9, t = i & 511;
  int len = enc_len[b];
  w[i] = (t < len) ? 1.0f / (float)len : 0.0f;
}

// x = bf16(concat(emb_W[ys_in[b,l]], ctx[b,:]))
__global__ void build_x(const float* __restrict__ emb_W, const int* __restrict__ ys_in,
                        const float* __restrict__ ctx, bf16_t* __restrict__ x_bf, int l) {
  int tid = blockIdx.x * 256 + threadIdx.x;      // B*1024
  if (tid >= NB * (NE + NOD)) return;
  int b = tid >> 10, j = tid & 1023;
  float v;
  if (j < NE) {
    int y = ys_in[b * NL + l];
    v = emb_W[(long)y * NE + j];
  } else {
    v = ctx[b * NOD + (j - NE)];
  }
  x_bf[tid] = (bf16_t)v;
}

__device__ __forceinline__ float sigf(float x) { return 1.0f / (1.0f + expf(-x)); }

__global__ void lstm_act(const float* __restrict__ gates, float* __restrict__ h,
                         float* __restrict__ c, bf16_t* __restrict__ h_bf,
                         bf16_t* __restrict__ z_bf) {
  int tid = blockIdx.x * 256 + threadIdx.x;      // B*H
  if (tid >= NB * NH) return;
  int b = tid >> 10, j = tid & 1023;
  const float* g = gates + (long)b * 4 * NH;
  float ig = sigf(g[j]);
  float fg = sigf(g[NH + j]);
  float gg = tanhf(g[2 * NH + j]);
  float og = sigf(g[3 * NH + j]);
  float cn = fg * c[tid] + ig * gg;
  float hn = og * tanhf(cn);
  c[tid] = cn;
  h[tid] = hn;
  h_bf[tid] = (bf16_t)hn;
  z_bf[(long)b * (NH + NOD) + j] = (bf16_t)hn;   // first half of [h;ctx]
}

// ---------------------------------------------------------------------------
// Location-aware attention: one workgroup per batch row (256 thr = 8 waves).
// ---------------------------------------------------------------------------
__global__ void att_kernel(const float* __restrict__ enc_pad,
                           const float* __restrict__ pre_enc,
                           const float* __restrict__ dec_t,
                           const float* __restrict__ conv_k,
                           const float* __restrict__ W_att,
                           const float* __restrict__ g_vec,
                           const float* __restrict__ W_o,
                           const float* __restrict__ b_o,
                           float* __restrict__ w,        // [B,T] in/out
                           float* __restrict__ ctx,      // [B,ODIM] out
                           bf16_t* __restrict__ z_bf,    // [B,1536]
                           float* __restrict__ ws_out,   // [B,L,T]
                           int l)
{
  __shared__ float wpad[NT + 2 * 100];   // 712
  __shared__ float ck[NC * NK];          // 2010
  __shared__ float convc[NC * NT];       // 5120
  __shared__ float dec[NATT];
  __shared__ float WattT[NC * NATT];     // transposed -> conflict-free
  __shared__ float ebuf[NT];
  __shared__ float ctxl[NENC];
  __shared__ float red[8];
  __shared__ float smax, ssum;

  const int b = blockIdx.x, tid = threadIdx.x;
  const int wv = tid >> 5, lane = tid & 31;

  for (int i = tid; i < NT + 200; i += 256) {
    int t = i - 100;
    wpad[i] = (t >= 0 && t < NT) ? w[b * NT + t] : 0.0f;
  }
  for (int i = tid; i < NC * NK; i += 256) ck[i] = conv_k[i];
  for (int i = tid; i < NATT; i += 256) dec[i] = dec_t[b * NATT + i];
  for (int i = tid; i < NC * NATT; i += 256) {
    int c = i / NATT, a = i % NATT;
    WattT[c * NATT + a] = W_att[a * NC + c];
  }
  __syncthreads();

  // 1D conv (correlation), padding=100
  for (int i = tid; i < NC * NT; i += 256) {
    int c = i >> 9, t = i & 511;
    const float* kk = ck + c * NK;
    const float* ww = wpad + t;
    float s = 0.f;
    for (int k = 0; k < NK; ++k) s += ww[k] * kk[k];
    convc[c * NT + t] = s;
  }
  __syncthreads();

  // e[t] = sum_a tanh(pre_enc + dec + att_conv) * g[a]   (one wave per t)
  for (int t = wv; t < NT; t += 8) {
    float acc = 0.f;
    const float* pe = pre_enc + ((long)(b * NT + t)) * NATT;
    for (int a = lane; a < NATT; a += 32) {
      float m = pe[a] + dec[a];
      float cc = 0.f;
#pragma unroll
      for (int c = 0; c < NC; ++c) cc += convc[c * NT + t] * WattT[c * NATT + a];
      acc += tanhf(m + cc) * g_vec[a];
    }
#pragma unroll
    for (int off = 16; off > 0; off >>= 1) acc += __shfl_down(acc, off, 32);
    if (lane == 0) ebuf[t] = acc;
  }
  __syncthreads();

  // softmax(2*e) over T
  float m = -INFINITY;
  for (int t = tid; t < NT; t += 256) m = fmaxf(m, ebuf[t]);
#pragma unroll
  for (int off = 16; off > 0; off >>= 1) m = fmaxf(m, __shfl_down(m, off, 32));
  if (lane == 0) red[wv] = m;
  __syncthreads();
  if (tid == 0) {
    float mm = red[0];
    for (int i = 1; i < 8; ++i) mm = fmaxf(mm, red[i]);
    smax = mm;
  }
  __syncthreads();
  float s = 0.f;
  for (int t = tid; t < NT; t += 256) {
    float ev = expf(2.0f * (ebuf[t] - smax));
    ebuf[t] = ev;
    s += ev;
  }
#pragma unroll
  for (int off = 16; off > 0; off >>= 1) s += __shfl_down(s, off, 32);
  if (lane == 0) red[wv] = s;
  __syncthreads();
  if (tid == 0) {
    float ss = 0.f;
    for (int i = 0; i < 8; ++i) ss += red[i];
    ssum = ss;
  }
  __syncthreads();
  const float inv = 1.0f / ssum;
  for (int t = tid; t < NT; t += 256) {
    float wn = ebuf[t] * inv;
    ebuf[t] = wn;
    w[b * NT + t] = wn;
    ws_out[((long)(b * NL + l)) * NT + t] = wn;
  }
  __syncthreads();

  // context = w_new @ enc_pad[b]   -> ctxl[ENC]
  for (int d = tid; d < NENC; d += 256) {
    float acc = 0.f;
    for (int t = 0; t < NT; ++t)
      acc += ebuf[t] * enc_pad[((long)(b * NT + t)) * NENC + d];
    ctxl[d] = acc;
  }
  __syncthreads();

  // ctx = ctxl @ W_o^T + b_o ; also fill second half of z
  for (int o = tid; o < NOD; o += 256) {
    const float* wo = W_o + (long)o * NENC;
    float acc = b_o[o];
    for (int d = 0; d < NENC; ++d) acc += ctxl[d] * wo[d];
    ctx[b * NOD + o] = acc;
    z_bf[(long)b * (NH + NOD) + NH + o] = (bf16_t)acc;
  }
}

// ---------------------------------------------------------------------------
// Final: log-softmax gather + argmax per (b,l)
// ---------------------------------------------------------------------------
__global__ void out_kernel(const float* __restrict__ logits, const int* __restrict__ ys_out,
                           float* __restrict__ lp_out, float* __restrict__ pred_out)
{
  __shared__ float redm[8];
  __shared__ int   redi[8];
  __shared__ float reds[8];
  __shared__ float gmax;
  __shared__ int   gidx;
  const int bl = blockIdx.x;                 // b*L + l
  const float* lg = logits + (long)bl * NV;
  const int tid = threadIdx.x, lane = tid & 31, wv = tid >> 5;

  float m = -INFINITY; int mi = 0x7fffffff;
  for (int v = tid; v < NV; v += 256) {
    float x = lg[v];
    if (x > m || (x == m && v < mi)) { m = x; mi = v; }
  }
#pragma unroll
  for (int off = 16; off > 0; off >>= 1) {
    float om = __shfl_down(m, off, 32);
    int   oi = __shfl_down(mi, off, 32);
    if (om > m || (om == m && oi < mi)) { m = om; mi = oi; }
  }
  if (lane == 0) { redm[wv] = m; redi[wv] = mi; }
  __syncthreads();
  if (tid == 0) {
    float mm = redm[0]; int ii = redi[0];
    for (int i = 1; i < 8; ++i)
      if (redm[i] > mm || (redm[i] == mm && redi[i] < ii)) { mm = redm[i]; ii = redi[i]; }
    gmax = mm; gidx = ii;
  }
  __syncthreads();
  float s = 0.f;
  for (int v = tid; v < NV; v += 256) s += expf(lg[v] - gmax);
#pragma unroll
  for (int off = 16; off > 0; off >>= 1) s += __shfl_down(s, off, 32);
  if (lane == 0) reds[wv] = s;
  __syncthreads();
  if (tid == 0) {
    float ss = 0.f;
    for (int i = 0; i < 8; ++i) ss += reds[i];
    int y = ys_out[bl];
    lp_out[bl] = lg[y] - gmax - logf(ss);
    pred_out[bl] = (float)gidx;
  }
}

// ---------------------------------------------------------------------------
extern "C" void kernel_launch(void* const* d_in, const int* in_sizes, int n_in,
                              void* d_out, int out_size, void* d_ws, size_t ws_size,
                              hipStream_t stream) {
  (void)in_sizes; (void)n_in; (void)out_size; (void)ws_size;
  const float* enc_pad = (const float*)d_in[0];
  const int*   enc_len = (const int*)d_in[1];
  const int*   ys_in   = (const int*)d_in[2];
  const int*   ys_out  = (const int*)d_in[3];
  const float* emb_W   = (const float*)d_in[4];
  const float* W_ih    = (const float*)d_in[5];
  const float* W_hh    = (const float*)d_in[6];
  const float* b_ih    = (const float*)d_in[7];
  const float* b_hh    = (const float*)d_in[8];
  const float* W_out   = (const float*)d_in[9];
  const float* b_out   = (const float*)d_in[10];
  const float* W_enc   = (const float*)d_in[11];
  const float* b_enc   = (const float*)d_in[12];
  const float* W_dec   = (const float*)d_in[13];
  const float* W_att   = (const float*)d_in[14];
  const float* conv_k  = (const float*)d_in[15];
  const float* g_vec   = (const float*)d_in[16];
  const float* W_o     = (const float*)d_in[17];
  const float* b_o     = (const float*)d_in[18];

  // workspace carve-up (256B aligned)
  char* wsp = (char*)d_ws;
  auto carve = [&](size_t bytes) -> char* {
    char* p = wsp;
    wsp += (bytes + 255) & ~(size_t)255;
    return p;
  };
  bf16_t* enc_bf  = (bf16_t*)carve((size_t)NB * NT * NENC * 2);
  bf16_t* wenc_bf = (bf16_t*)carve((size_t)NATT * NENC * 2);
  bf16_t* wih_bf  = (bf16_t*)carve((size_t)4 * NH * (NE + NOD) * 2);
  bf16_t* whh_bf  = (bf16_t*)carve((size_t)4 * NH * NH * 2);
  bf16_t* wout_bf = (bf16_t*)carve((size_t)NVPAD * (NH + NOD) * 2);
  bf16_t* wdec_bf = (bf16_t*)carve((size_t)NATT * NH * 2);
  float*  pre_enc = (float*)carve((size_t)NB * NT * NATT * 4);
  float*  gates   = (float*)carve((size_t)NB * 4 * NH * 4);
  float*  dec_t   = (float*)carve((size_t)NB * NATT * 4);
  float*  states  = (float*)carve((size_t)(NB * NH * 2 + NB * NOD) * 4);
  float*  h    = states;
  float*  cst  = states + NB * NH;
  float*  ctx  = states + NB * NH * 2;
  float*  wbuf = (float*)carve((size_t)NB * NT * 4);
  bf16_t* x_bf = (bf16_t*)carve((size_t)NB * (NE + NOD) * 2);
  bf16_t* h_bf = (bf16_t*)carve((size_t)NB * NH * 2);
  bf16_t* z_bf = (bf16_t*)carve((size_t)NB * (NH + NOD) * 2);
  float*  bcomb = (float*)carve((size_t)4 * NH * 4);

  float* out      = (float*)d_out;
  float* o_logits = out;                                     // [B,L,V]
  float* o_lp     = out + (long)NB * NL * NV;                // [B,L]
  float* o_pred   = o_lp + NB * NL;                          // [B,L]
  float* o_ws     = o_pred + NB * NL;                        // [B,L,T]

  const dim3 blk(256);
  auto gcvt = [](long n) { long b = (n + 255) / 256; return dim3((unsigned)(b > 4096 ? 4096 : b)); };

  // 1) one-time conversions
  cvt_f32_bf16<<<gcvt((long)NB * NT * NENC), blk, 0, stream>>>(enc_pad, enc_bf, (long)NB * NT * NENC);
  cvt_f32_bf16<<<gcvt((long)NATT * NENC), blk, 0, stream>>>(W_enc, wenc_bf, (long)NATT * NENC);
  cvt_f32_bf16<<<gcvt((long)4 * NH * (NE + NOD)), blk, 0, stream>>>(W_ih, wih_bf, (long)4 * NH * (NE + NOD));
  cvt_f32_bf16<<<gcvt((long)4 * NH * NH), blk, 0, stream>>>(W_hh, whh_bf, (long)4 * NH * NH);
  cvt_f32_bf16<<<gcvt((long)NATT * NH), blk, 0, stream>>>(W_dec, wdec_bf, (long)NATT * NH);
  cvt_wout_pad<<<gcvt((long)NVPAD * (NH + NOD)), blk, 0, stream>>>(W_out, wout_bf);
  bias_comb<<<dim3(16), blk, 0, stream>>>(b_ih, b_hh, bcomb);

  // 2) init states + w0
  zero_f32<<<gcvt(NB * NH * 2 + NB * NOD), blk, 0, stream>>>(states, NB * NH * 2 + NB * NOD);
  init_w<<<dim3((NB * NT + 255) / 256), blk, 0, stream>>>(enc_len, wbuf);

  // 3) pre_enc = enc_pad @ W_enc^T + b_enc  (M=16384, N=512, K=512)
  gemm32_bf16<<<dim3(4, (NB * NT) / 32), blk, 0, stream>>>(
      enc_bf, wenc_bf, NENC, NENC, NENC,
      nullptr, nullptr, 0, 0, 0,
      b_enc, pre_enc, NATT, NATT, NATT / 16);

  // 4) decode loop
  for (int l = 0; l < NL; ++l) {
    build_x<<<dim3((NB * (NE + NOD) + 255) / 256), blk, 0, stream>>>(emb_W, ys_in, ctx, x_bf, l);

    // gates = x @ W_ih^T + h @ W_hh^T + (b_ih+b_hh)   (M=32, N=4096, K=1024+1024)
    gemm32_bf16<<<dim3(32, 1), blk, 0, stream>>>(
        x_bf, wih_bf, NE + NOD, NE + NOD, NE + NOD,
        h_bf, whh_bf, NH, NH, NH,
        bcomb, gates, 4 * NH, 4 * NH, (4 * NH) / 16);

    lstm_act<<<dim3((NB * NH + 255) / 256), blk, 0, stream>>>(gates, h, cst, h_bf, z_bf);

    // dec_t = h @ W_dec^T   (M=32, N=512, K=1024)
    gemm32_bf16<<<dim3(4, 1), blk, 0, stream>>>(
        h_bf, wdec_bf, NH, NH, NH,
        nullptr, nullptr, 0, 0, 0,
        nullptr, dec_t, NATT, NATT, NATT / 16);

    att_kernel<<<dim3(NB), blk, 0, stream>>>(enc_pad, pre_enc, dec_t, conv_k, W_att,
                                             g_vec, W_o, b_o, wbuf, ctx, z_bf, o_ws, l);

    // logits[:, l, :] = [h;ctx] @ W_out^T + b_out  (M=32, N=1000(pad 1008), K=1536)
    gemm32_bf16<<<dim3(8, 1), blk, 0, stream>>>(
        z_bf, wout_bf, NH + NOD, NH + NOD, NH + NOD,
        nullptr, nullptr, 0, 0, 0,
        b_out, o_logits + (long)l * NV, (long)NL * NV, NV, NVPAD / 16);
  }

  // 5) log-softmax gather + argmax
  out_kernel<<<dim3(NB * NL), blk, 0, stream>>>(o_logits, ys_out, o_lp, o_pred);
}
